// WeaveLayer_76063870812664
// MI455X (gfx1250) — matrix operand order repaired and checked by reading, not compiled, not run on hardware
//
#include <hip/hip_runtime.h>
#include <hip/hip_bf16.h>

#define N_ATOMS 100000
#define N_PAIRS 1600000
#define D_ATOM 75
#define D_PAIR 14
#define HDIM 50

typedef __attribute__((ext_vector_type(16))) __bf16 v16bf;
typedef __attribute__((ext_vector_type(8)))  float  v8f;

// float -> bf16 bits, round-to-nearest-even
__device__ __forceinline__ unsigned short f2bf(float f) {
  unsigned u = __float_as_uint(f);
  unsigned r = ((u >> 16) & 1u) + 0x7fffu;
  return (unsigned short)((u + r) >> 16);
}

union Frag { uint4 q[2]; v16bf v; };

// A-matrix 16x32 bf16 fragment (ISA 7.12.2): lane half koff 0/8,
// elems 0-7 -> K = ks*32+koff+0..7, elems 8-15 -> K = ks*32+16+koff+0..7
__device__ __forceinline__ v16bf load_fragA(const unsigned short* row, int ks, int laneHi) {
  Frag f;
  int koff = laneHi * 8;
  f.q[0] = *(const uint4*)(row + ks * 32 + koff);
  f.q[1] = *(const uint4*)(row + ks * 32 + 16 + koff);
  return f.v;
}
// B-matrix 32x16 bf16 fragment: lane = column, same K split as A (koff 0/8, +16)
__device__ __forceinline__ v16bf load_fragB(const unsigned short* row, int ks, int laneHi) {
  Frag f;
  int base = ks * 32 + laneHi * 8;
  f.q[0] = *(const uint4*)(row + base);
  f.q[1] = *(const uint4*)(row + base + 16);
  return f.v;
}

// One wave computes a 16x64 output tile: X[16x128 bf16 in LDS] @ Wt[64x128 bf16 in LDS]^T
// K padded to 128, N padded to 64; bias + relu epilogue, only cols < 50 stored.
__device__ __forceinline__ void wmma_16x64_store(
    const unsigned short* Htile, const unsigned short* Wt,
    const float* __restrict__ bias, float* __restrict__ outRow0, int lane)
{
  int laneHi = lane >> 4;
  int r16 = lane & 15;
  const unsigned short* Arow = Htile + r16 * 128;
  v8f acc[4];
#pragma unroll
  for (int nt = 0; nt < 4; ++nt)
#pragma unroll
    for (int e = 0; e < 8; ++e) acc[nt][e] = 0.f;
#pragma unroll
  for (int ks = 0; ks < 4; ++ks) {
    v16bf a = load_fragA(Arow, ks, laneHi);
#pragma unroll
    for (int nt = 0; nt < 4; ++nt) {
      v16bf b = load_fragB(Wt + (nt * 16 + r16) * 128, ks, laneHi);
      acc[nt] = __builtin_amdgcn_wmma_f32_16x16x32_bf16(
          false, a, false, b, (short)0, acc[nt], false, false);
    }
  }
  int rbase = laneHi * 8;
#pragma unroll
  for (int nt = 0; nt < 4; ++nt) {
    int col = nt * 16 + r16;
    if (col < HDIM) {
      float bv = bias[col];
#pragma unroll
      for (int r = 0; r < 8; ++r) {
        float v = acc[nt][r] + bv;
        outRow0[(size_t)(rbase + r) * HDIM + col] = v > 0.f ? v : 0.f;
      }
    }
  }
}

// ---- K_w: convert W_A, W_P (f32 [100][50]) to transposed padded bf16 [64][128]
__global__ void __launch_bounds__(256) weight_prep_kernel(
    const float* __restrict__ W_A, const float* __restrict__ W_P,
    unsigned short* __restrict__ WaT, unsigned short* __restrict__ WpT)
{
  int idx = blockIdx.x * 256 + threadIdx.x;   // 0..16383
  int which = idx >> 13;
  int rem = idx & 8191;
  int n = rem >> 7, k = rem & 127;
  const float* W = which ? W_P : W_A;
  unsigned short* dst = which ? WpT : WaT;
  dst[n * 128 + k] = (n < HDIM && k < 100) ? f2bf(W[k * HDIM + n]) : (unsigned short)0;
}

// ---- K1: per-atom linears: AA (relu) -> AH[:,0:50], U = x@Wtop + b_AP, V = x@Wbot
__global__ void __launch_bounds__(256) atom_linear_kernel(
    const float* __restrict__ AF,
    const float* __restrict__ W_AA, const float* __restrict__ b_AA,
    const float* __restrict__ W_AP, const float* __restrict__ b_AP,
    float* __restrict__ AH, float* __restrict__ U, float* __restrict__ V)
{
  int idx = blockIdx.x * 256 + threadIdx.x;
  if (idx >= N_ATOMS * 150) return;
  int a = idx / 150, o = idx % 150;
  const float* x = AF + (size_t)a * D_ATOM;
  float acc;
  if (o < 50) {
    acc = b_AA[o];
    for (int k = 0; k < D_ATOM; ++k) acc += x[k] * W_AA[k * HDIM + o];
    AH[(size_t)a * 100 + o] = acc > 0.f ? acc : 0.f;
  } else if (o < 100) {
    int h = o - 50;
    acc = b_AP[h];                                  // fold bias into U once
    for (int k = 0; k < D_ATOM; ++k) acc += x[k] * W_AP[k * HDIM + h];
    U[(size_t)a * HDIM + h] = acc;
  } else {
    int h = o - 100;
    acc = 0.f;
    for (int k = 0; k < D_ATOM; ++k) acc += x[k] * W_AP[(D_ATOM + k) * HDIM + h];
    V[(size_t)a * HDIM + h] = acc;
  }
}

// ---- K2: segment-sum of relu(pair@W_PA+b) without atomics (pair_split is sorted:
// each atom's pairs are a contiguous range found by binary search). Wave per atom.
__global__ void __launch_bounds__(256) pair_atom_kernel(
    const float* __restrict__ PF, const int* __restrict__ pair_split,
    const float* __restrict__ W_PA, const float* __restrict__ b_PA,
    float* __restrict__ AH)
{
  int wave = threadIdx.x >> 5, lane = threadIdx.x & 31;
  int a = blockIdx.x * 8 + wave;
  int lo = 0, hi = N_PAIRS;
  while (lo < hi) { int mid = (lo + hi) >> 1; if (pair_split[mid] <  a) lo = mid + 1; else hi = mid; }
  int lo2 = lo, hi2 = N_PAIRS;
  while (lo2 < hi2) { int mid = (lo2 + hi2) >> 1; if (pair_split[mid] < a + 1) lo2 = mid + 1; else hi2 = mid; }
  int h0 = lane, h1 = lane + 32;
  float w0[D_PAIR], w1[D_PAIR];
#pragma unroll
  for (int k = 0; k < D_PAIR; ++k) {
    w0[k] = W_PA[k * HDIM + h0];
    w1[k] = (h1 < HDIM) ? W_PA[k * HDIM + h1] : 0.f;
  }
  float bb0 = b_PA[h0];
  float bb1 = (h1 < HDIM) ? b_PA[h1] : 0.f;
  float acc0 = 0.f, acc1 = 0.f;
  for (int p = lo; p < lo2; ++p) {
    const float* pf = PF + (size_t)p * D_PAIR;
    float d0 = bb0, d1 = bb1;
#pragma unroll
    for (int k = 0; k < D_PAIR; ++k) { float x = pf[k]; d0 += x * w0[k]; d1 += x * w1[k]; }
    acc0 += d0 > 0.f ? d0 : 0.f;
    acc1 += d1 > 0.f ? d1 : 0.f;
  }
  AH[(size_t)a * 100 + 50 + h0] = acc0;
  if (h1 < HDIM) AH[(size_t)a * 100 + 50 + h1] = acc1;
}

// ---- K3: A = relu([AA|PA] @ W_A + b_A) via WMMA. Wave per 16-atom tile.
__global__ void __launch_bounds__(256) atom_out_kernel(
    const float* __restrict__ AH, const unsigned short* __restrict__ WaT,
    const float* __restrict__ b_A, float* __restrict__ Aout)
{
  __shared__ __align__(16) unsigned short sW[64 * 128];
  __shared__ __align__(16) unsigned short sH[8 * 16 * 128];
  int tid = threadIdx.x;
  for (int idx = tid; idx < 1024; idx += 256)
    ((uint4*)sW)[idx] = ((const uint4*)WaT)[idx];
  __syncthreads();
  int wave = tid >> 5, lane = tid & 31;
  int tile = blockIdx.x * 8 + wave;
  bool valid = tile < (N_ATOMS / 16);
  int t = lane >> 1, hh = lane & 1;
  if (valid) {
    int a = tile * 16 + t;
    unsigned short* Hrow = sH + wave * 2048 + t * 128;
    const float* src = AH + (size_t)a * 100 + hh * 50;
    for (int m = 0; m < 50; ++m) Hrow[hh * 50 + m] = f2bf(src[m]);
#pragma unroll
    for (int m = 0; m < 14; ++m) Hrow[100 + hh * 14 + m] = 0;   // K pad 100->128
  }
  __syncthreads();
  if (valid)
    wmma_16x64_store(sH + wave * 2048, sW, b_A, Aout + (size_t)tile * 16 * HDIM, lane);
}

// ---- K4: P = relu([relu(U_i+V_j)+relu(U_j+V_i) | relu(pf@W_PP+b)] @ W_P + b_P)
// Wave per 16-pair tile; fused PP GEMV; WMMA over K=128.
__global__ void __launch_bounds__(256) pair_out_kernel(
    const float* __restrict__ PF, const int* __restrict__ A2P,
    const float* __restrict__ U, const float* __restrict__ V,
    const unsigned short* __restrict__ WpT,
    const float* __restrict__ W_PP, const float* __restrict__ b_PP,
    const float* __restrict__ b_P, float* __restrict__ Pout)
{
  __shared__ __align__(16) unsigned short sW[64 * 128];
  __shared__ float sWPP[D_PAIR * HDIM];
  __shared__ float sbPP[HDIM];
  __shared__ __align__(16) unsigned short sH[8 * 16 * 128];

  int tid = threadIdx.x;
  for (int idx = tid; idx < 1024; idx += 256)
    ((uint4*)sW)[idx] = ((const uint4*)WpT)[idx];
  for (int idx = tid; idx < D_PAIR * HDIM; idx += 256) sWPP[idx] = W_PP[idx];
  if (tid < HDIM) sbPP[tid] = b_PP[tid];
  __syncthreads();

  int wave = tid >> 5, lane = tid & 31;
  int tile = blockIdx.x * 8 + wave;
  int p0 = tile * 16;
  int t = lane >> 1, hh = lane & 1;              // 2 lanes per pair, 25 h each
  int p = p0 + t;
  unsigned short* Hrow = sH + wave * 2048 + t * 128;
  int i = A2P[2 * p], j = A2P[2 * p + 1];
  float pf[D_PAIR];
#pragma unroll
  for (int k = 0; k < D_PAIR; ++k) pf[k] = PF[(size_t)p * D_PAIR + k];
  const float* Ui = U + (size_t)i * HDIM; const float* Vi = V + (size_t)i * HDIM;
  const float* Uj = U + (size_t)j * HDIM; const float* Vj = V + (size_t)j * HDIM;
  for (int m = 0; m < 25; ++m) {
    int h = hh * 25 + m;
    float s1 = Ui[h] + Vj[h];                    // b_AP already folded into U
    float s2 = Uj[h] + Vi[h];
    Hrow[h] = f2bf((s1 > 0.f ? s1 : 0.f) + (s2 > 0.f ? s2 : 0.f));
    float acc = sbPP[h];
#pragma unroll
    for (int k = 0; k < D_PAIR; ++k) acc += pf[k] * sWPP[k * HDIM + h];
    Hrow[50 + h] = f2bf(acc > 0.f ? acc : 0.f);
  }
#pragma unroll
  for (int m = 0; m < 14; ++m) Hrow[100 + hh * 14 + m] = 0;   // K pad
  __syncthreads();

  wmma_16x64_store(sH + wave * 2048, sW, b_P, Pout + (size_t)p0 * HDIM, lane);
}

extern "C" void kernel_launch(void* const* d_in, const int* in_sizes, int n_in,
                              void* d_out, int out_size, void* d_ws, size_t ws_size,
                              hipStream_t stream)
{
  const float* AF   = (const float*)d_in[0];
  const float* PF   = (const float*)d_in[1];
  const int*   PS   = (const int*)d_in[2];
  const int*   A2P  = (const int*)d_in[3];
  const float* W_AA = (const float*)d_in[4];
  const float* b_AA = (const float*)d_in[5];
  const float* W_PA = (const float*)d_in[6];
  const float* b_PA = (const float*)d_in[7];
  const float* W_A  = (const float*)d_in[8];
  const float* b_A  = (const float*)d_in[9];
  const float* W_AP = (const float*)d_in[10];
  const float* b_AP = (const float*)d_in[11];
  const float* W_PP = (const float*)d_in[12];
  const float* b_PP = (const float*)d_in[13];
  const float* W_P  = (const float*)d_in[14];
  const float* b_P  = (const float*)d_in[15];

  // workspace: AH[N_ATOMS][100] (AA|PA), U/V[N_ATOMS][50], two bf16 weight tiles
  float* AH = (float*)d_ws;
  float* U  = AH + (size_t)N_ATOMS * 100;
  float* V  = U + (size_t)N_ATOMS * HDIM;
  unsigned short* WaT = (unsigned short*)(V + (size_t)N_ATOMS * HDIM);
  unsigned short* WpT = WaT + 64 * 128;

  float* Aout = (float*)d_out;
  float* Pout = Aout + (size_t)N_ATOMS * HDIM;

  weight_prep_kernel<<<64, 256, 0, stream>>>(W_A, W_P, WaT, WpT);
  atom_linear_kernel<<<(N_ATOMS * 150 + 255) / 256, 256, 0, stream>>>(
      AF, W_AA, b_AA, W_AP, b_AP, AH, U, V);
  pair_atom_kernel<<<N_ATOMS / 8, 256, 0, stream>>>(PF, PS, W_PA, b_PA, AH);
  atom_out_kernel<<<(N_ATOMS / 16 + 7) / 8, 256, 0, stream>>>(AH, WaT, b_A, Aout);
  pair_out_kernel<<<N_PAIRS / 128, 256, 0, stream>>>(
      PF, A2P, U, V, WpT, W_PP, b_PP, b_P, Pout);
}
